// GNNEncoder_19061064860125
// MI455X (gfx1250) — matrix operand order repaired and checked by reading, not compiled, not run on hardware
//
#include <hip/hip_runtime.h>
#include <math.h>

#define NA 20000
#define NP 40000
#define HIDDEN 256
#define NHEADS 8
#define DHEAD 32
#define E_W 60000
#define E_RW 60000
#define E_C 120000
#define E_TOT (E_W + E_RW + E_C)

typedef __attribute__((ext_vector_type(16))) __bf16 v16bf;
typedef __attribute__((ext_vector_type(8)))  float  v8f;
typedef __attribute__((ext_vector_type(8)))  unsigned short us8;

union BFrag { unsigned short u[16]; v16bf v; };

__device__ __forceinline__ unsigned short f2bf(float f) {
    unsigned int u = __float_as_uint(f);
    u += 0x7fffu + ((u >> 16) & 1u);   // round-to-nearest-even
    return (unsigned short)(u >> 16);
}
__device__ __forceinline__ float bf2f(unsigned short h) {
    return __uint_as_float(((unsigned int)h) << 16);
}
__device__ __forceinline__ void atomicMaxF(float* addr, float val) {
    if (val >= 0.0f) atomicMax((int*)addr, __float_as_int(val));
    else             atomicMin((unsigned int*)addr, __float_as_uint(val));
}

// ---------------------------------------------------------------------------
// D[M,N] = act(bf16(A[M,K]) @ bf16(Wt[N,K])^T + bias)
// One wave computes a 16x64 tile (4 accumulators, shared A fragment).
// K loop is software-pipelined with double-buffered fragments so the next
// k-step's loads are in flight while the current WMMAs execute.
// ---------------------------------------------------------------------------
__global__ void k_gemm(const unsigned short* __restrict__ A,
                       const unsigned short* __restrict__ Wt,
                       const float* __restrict__ bias,
                       float* __restrict__ C,              // optional f32 out
                       unsigned short* __restrict__ Cbf,   // optional bf16 out
                       int M, int K, int N, int relu) {
    int gw   = (int)((blockIdx.x * blockDim.x + threadIdx.x) >> 5);
    int lane = threadIdx.x & 31;
    int ncg  = N >> 6;                       // 64-wide column groups
    int total = (M >> 4) * ncg;
    if (gw >= total) return;
    int tm = gw / ncg, cg = gw - tm * ncg;
    int l  = lane & 15;
    int hi = lane >> 4;
    int row = (tm << 4) + l;

    v8f acc[4];
#pragma unroll
    for (int j = 0; j < 4; ++j) acc[j] = (v8f){0.f,0.f,0.f,0.f,0.f,0.f,0.f,0.f};

    const unsigned short* ap = A + (size_t)row * K + (hi << 3);
    const unsigned short* bp[4];
#pragma unroll
    for (int j = 0; j < 4; ++j)
        bp[j] = Wt + (size_t)((cg << 6) + (j << 4) + l) * K + (hi << 3);

    // prologue: fragments for k0 = 0
    BFrag fa, fb[4];
    *(us8*)&fa.u[0] = *(const us8*)(ap);
    *(us8*)&fa.u[8] = *(const us8*)(ap + 16);
#pragma unroll
    for (int j = 0; j < 4; ++j) {
        *(us8*)&fb[j].u[0] = *(const us8*)(bp[j]);
        *(us8*)&fb[j].u[8] = *(const us8*)(bp[j] + 16);
    }
    // steady state: prefetch k0+32 while multiplying k0
    for (int k0 = 32; k0 < K; k0 += 32) {
        BFrag na, nb[4];
        *(us8*)&na.u[0] = *(const us8*)(ap + k0);
        *(us8*)&na.u[8] = *(const us8*)(ap + k0 + 16);
#pragma unroll
        for (int j = 0; j < 4; ++j) {
            *(us8*)&nb[j].u[0] = *(const us8*)(bp[j] + k0);
            *(us8*)&nb[j].u[8] = *(const us8*)(bp[j] + k0 + 16);
        }
#pragma unroll
        for (int j = 0; j < 4; ++j)
            acc[j] = __builtin_amdgcn_wmma_f32_16x16x32_bf16(false, fa.v, false, fb[j].v,
                                                             (short)0, acc[j], false, false);
        fa = na;
#pragma unroll
        for (int j = 0; j < 4; ++j) fb[j] = nb[j];
    }
    // epilogue k-step
#pragma unroll
    for (int j = 0; j < 4; ++j)
        acc[j] = __builtin_amdgcn_wmma_f32_16x16x32_bf16(false, fa.v, false, fb[j].v,
                                                         (short)0, acc[j], false, false);

    int rb = (tm << 4) + (hi << 3);
#pragma unroll
    for (int j = 0; j < 4; ++j) {
        int col = (cg << 6) + (j << 4) + l;
        float bv = bias ? bias[col] : 0.0f;
#pragma unroll
        for (int r = 0; r < 8; ++r) {
            float x = acc[j][r] + bv;
            if (relu) x = fmaxf(x, 0.0f);
            if (C)   C[(size_t)(rb + r) * N + col] = x;
            if (Cbf) Cbf[(size_t)(rb + r) * N + col] = f2bf(x);
        }
    }
}

// ---------------------------------------------------------------------------
// Y[e,h,f] = sum_d X[src[e], h*32+d] * Rel[h,d,f]  (RelT stored [h][f][d] bf16)
// One wave per (16 edges, head): gathered A fragment loaded once, feeds the
// two 16-column WMMAs (f=0..15, 16..31).
// ---------------------------------------------------------------------------
__global__ void k_edge(const unsigned short* __restrict__ Xbf,
                       const int* __restrict__ src,
                       const unsigned short* __restrict__ RelT,
                       unsigned short* __restrict__ Y,
                       int E) {
    int gw   = (int)((blockIdx.x * blockDim.x + threadIdx.x) >> 5);
    int lane = threadIdx.x & 31;
    int total = (E >> 4) * NHEADS;
    if (gw >= total) return;
    int te = gw / NHEADS;
    int h  = gw - te * NHEADS;
    int l  = lane & 15, hi = lane >> 4;
    int er = (te << 4) + l;
    int s  = src[er];

    BFrag fa;
    const unsigned short* ap = Xbf + (size_t)s * HIDDEN + h * DHEAD + (hi << 3);
    *(us8*)&fa.u[0] = *(const us8*)ap;
    *(us8*)&fa.u[8] = *(const us8*)(ap + 16);

    v8f acc[2];
#pragma unroll
    for (int j = 0; j < 2; ++j) {
        BFrag fb;
        const unsigned short* bp = RelT + h * (DHEAD * DHEAD) + ((j << 4) + l) * DHEAD + (hi << 3);
        *(us8*)&fb.u[0] = *(const us8*)bp;
        *(us8*)&fb.u[8] = *(const us8*)(bp + 16);
        acc[j] = (v8f){0.f,0.f,0.f,0.f,0.f,0.f,0.f,0.f};
        acc[j] = __builtin_amdgcn_wmma_f32_16x16x32_bf16(false, fa.v, false, fb.v,
                                                         (short)0, acc[j], false, false);
    }
    int eb = (te << 4) + (hi << 3);
#pragma unroll
    for (int j = 0; j < 2; ++j)
#pragma unroll
        for (int r = 0; r < 8; ++r)
            Y[((size_t)(eb + r) * NHEADS + h) * DHEAD + (j << 4) + l] = f2bf(acc[j][r]);
}

// ---- elementwise / segment kernels ----------------------------------------
__global__ void k_transpose_bf16(const float* __restrict__ in,
                                 unsigned short* __restrict__ out,
                                 int din, int dout, int batch) {
    int idx = blockIdx.x * blockDim.x + threadIdx.x;
    int per = din * dout;
    if (idx >= batch * per) return;
    int b = idx / per;
    int r = idx - b * per;
    int j = r / din;       // dout index
    int i = r - j * din;   // din index
    out[idx] = f2bf(in[(size_t)b * per + (size_t)i * dout + j]);
}

__global__ void k_to_bf16(const float* __restrict__ in, unsigned short* __restrict__ out, int n) {
    int i = blockIdx.x * blockDim.x + threadIdx.x;
    if (i < n) out[i] = f2bf(in[i]);
}

__global__ void k_fill(float* p, float v, int n) {
    int i = blockIdx.x * blockDim.x + threadIdx.x;
    if (i < n) p[i] = v;
}

__global__ void k_alpha(const unsigned short* __restrict__ ke, const float* __restrict__ q,
                        const int* __restrict__ dst, const float* __restrict__ prel,
                        float* __restrict__ alpha, float* __restrict__ mbuf, int E) {
    int idx = blockIdx.x * blockDim.x + threadIdx.x;
    if (idx >= E * NHEADS) return;
    int e = idx >> 3, h = idx & 7;
    int d = dst[e];
    const unsigned short* kp = ke + ((size_t)e * NHEADS + h) * DHEAD;
    const float* qp = q + (size_t)d * HIDDEN + h * DHEAD;
    float s = 0.f;
#pragma unroll
    for (int f = 0; f < DHEAD; ++f) s += bf2f(kp[f]) * qp[f];
    float a = s * prel[h] * 0.17677669529663687f;   // 1/sqrt(32)
    alpha[idx] = a;
    atomicMaxF(&mbuf[(size_t)d * NHEADS + h], a);
}

__global__ void k_expsum(const float* __restrict__ mbuf, const int* __restrict__ dst,
                         float* __restrict__ alpha, float* __restrict__ sbuf, int E) {
    int idx = blockIdx.x * blockDim.x + threadIdx.x;
    if (idx >= E * NHEADS) return;
    int e = idx >> 3, h = idx & 7;
    int d = dst[e];
    float m = mbuf[(size_t)d * NHEADS + h];
    if (!isfinite(m)) m = 0.0f;
    float ex = expf(alpha[idx] - m);
    alpha[idx] = ex;
    atomicAdd(&sbuf[(size_t)d * NHEADS + h], ex);
}

__global__ void k_scatter(const unsigned short* __restrict__ ve, const int* __restrict__ dst,
                          const float* __restrict__ alpha, const float* __restrict__ sbuf,
                          float* __restrict__ agg, int E) {
    int idx = blockIdx.x * blockDim.x + threadIdx.x;
    if (idx >= E * NHEADS) return;
    int e = idx >> 3, h = idx & 7;
    int d = dst[e];
    float w = alpha[idx] / (sbuf[(size_t)d * NHEADS + h] + 1e-16f);
    const unsigned short* vp = ve + ((size_t)e * NHEADS + h) * DHEAD;
    float* ap = agg + (size_t)d * HIDDEN + h * DHEAD;
#pragma unroll
    for (int f = 0; f < DHEAD; ++f) atomicAdd(&ap[f], w * bf2f(vp[f]));
}

__global__ void k_gelu_bf16(const float* __restrict__ in, unsigned short* __restrict__ out, int n) {
    int i = blockIdx.x * blockDim.x + threadIdx.x;
    if (i >= n) return;
    float x = in[i];
    float g = 0.5f * x * (1.0f + erff(x * 0.70710678118654752f));
    out[i] = f2bf(g);
}

__global__ void k_mix(const float* __restrict__ o, const float* __restrict__ skip,
                      float* __restrict__ h, int n) {
    int i = blockIdx.x * blockDim.x + threadIdx.x;
    if (i >= n) return;
    float sk = 1.0f / (1.0f + expf(-skip[0]));
    h[i] = sk * o[i] + (1.0f - sk) * h[i];
}

__global__ void k_bn_stats(const float* __restrict__ x, int rows, float* __restrict__ sums) {
    int col = blockIdx.x;  // 0..255
    float s = 0.f, s2 = 0.f;
    for (int r = threadIdx.x; r < rows; r += blockDim.x) {
        float v = x[(size_t)r * HIDDEN + col];
        s += v; s2 += v * v;
    }
    __shared__ float sh[256], sh2[256];
    sh[threadIdx.x] = s; sh2[threadIdx.x] = s2;
    __syncthreads();
    for (int off = 128; off > 0; off >>= 1) {
        if ((int)threadIdx.x < off) { sh[threadIdx.x] += sh[threadIdx.x + off];
                                      sh2[threadIdx.x] += sh2[threadIdx.x + off]; }
        __syncthreads();
    }
    if (threadIdx.x == 0) { sums[col] = sh[0]; sums[256 + col] = sh2[0]; }
}

__global__ void k_bn_apply(float* __restrict__ x, const float* __restrict__ sums,
                           const float* __restrict__ g, const float* __restrict__ b, int rows) {
    int idx = blockIdx.x * blockDim.x + threadIdx.x;
    if (idx >= rows * HIDDEN) return;
    int col = idx & 255;
    float inv = 1.0f / (float)rows;
    float mu  = sums[col] * inv;
    float var = sums[256 + col] * inv - mu * mu;
    x[idx] = (x[idx] - mu) * rsqrtf(var + 1e-5f) * g[col] + b[col];
}

__global__ void k_copy(const float* __restrict__ in, float* __restrict__ out, int n) {
    int i = blockIdx.x * blockDim.x + threadIdx.x;
    if (i < n) out[i] = in[i];
}

// ---------------------------------------------------------------------------
static inline unsigned div_up(size_t a, size_t b) { return (unsigned)((a + b - 1) / b); }

extern "C" void kernel_launch(void* const* d_in, const int* in_sizes, int n_in,
                              void* d_out, int out_size, void* d_ws, size_t ws_size,
                              hipStream_t stream) {
    (void)in_sizes; (void)n_in; (void)out_size; (void)ws_size;
    const float* xa  = (const float*)d_in[0];
    const float* xp  = (const float*)d_in[1];
    const int*   ew  = (const int*)d_in[2];
    const int*   erw = (const int*)d_in[3];
    const int*   ec  = (const int*)d_in[4];
    auto Pf = [&](int i) -> const float* { return (const float*)d_in[i]; };

    // --- workspace layout -------------------------------------------------
    char* wsb = (char*)d_ws;
    size_t cur = 0;
    auto alF = [&](size_t n) -> float* {
        float* p = (float*)(wsb + cur); cur = (cur + n * 4 + 255) & ~(size_t)255; return p; };
    auto alH = [&](size_t n) -> unsigned short* {
        unsigned short* p = (unsigned short*)(wsb + cur); cur = (cur + n * 2 + 255) & ~(size_t)255; return p; };

    float* h_a   = alF((size_t)NA * HIDDEN);
    float* h_p   = alF((size_t)NP * HIDDEN);
    float* q_a   = alF((size_t)NA * HIDDEN);
    float* q_p   = alF((size_t)NP * HIDDEN);
    float* o_a   = alF((size_t)NA * HIDDEN);
    float* o_p   = alF((size_t)NP * HIDDEN);
    float* agg_a = alF((size_t)NA * HIDDEN);
    float* agg_p = alF((size_t)NP * HIDDEN);
    float* alpha = alF((size_t)E_TOT * NHEADS);
    float* m_a = alF((size_t)NA * NHEADS);
    float* s_a = alF((size_t)NA * NHEADS);
    float* m_p = alF((size_t)NP * NHEADS);
    float* s_p = alF((size_t)NP * NHEADS);
    float* bns = alF(512);

    unsigned short* xb_a = alH((size_t)NA * 128);
    unsigned short* xb_p = alH((size_t)NP * 128);
    unsigned short* hb_a = alH((size_t)NA * HIDDEN);
    unsigned short* hb_p = alH((size_t)NP * HIDDEN);
    unsigned short* kb_a = alH((size_t)NA * HIDDEN);
    unsigned short* kb_p = alH((size_t)NP * HIDDEN);
    unsigned short* vb_a = alH((size_t)NA * HIDDEN);
    unsigned short* vb_p = alH((size_t)NP * HIDDEN);
    unsigned short* gb_a = alH((size_t)NA * HIDDEN);
    unsigned short* gb_p = alH((size_t)NP * HIDDEN);
    unsigned short* keb  = alH((size_t)E_C * HIDDEN);    // reused per relation
    unsigned short* veb  = alH((size_t)E_TOT * HIDDEN);
    unsigned short* kt_a = alH(65536); unsigned short* kt_p = alH(65536);
    unsigned short* qt_a = alH(65536); unsigned short* qt_p = alH(65536);
    unsigned short* vt_a = alH(65536); unsigned short* vt_p = alH(65536);
    unsigned short* ot_a = alH(65536); unsigned short* ot_p = alH(65536);
    unsigned short* art[3]; unsigned short* mrt[3];
    for (int r = 0; r < 3; ++r) { art[r] = alH(8192); mrt[r] = alH(8192); }
    unsigned short* lt_a = alH(32768); unsigned short* lt_p = alH(32768);

    auto gemm = [&](const unsigned short* A, const unsigned short* Wt, const float* bias,
                    float* C, unsigned short* Cbf, int M, int K, int N, int relu) {
        unsigned blk = div_up((size_t)(M / 16) * (N / 64), 8);
        k_gemm<<<blk, 256, 0, stream>>>(A, Wt, bias, C, Cbf, M, K, N, relu);
    };
    auto T = [&](const float* in, unsigned short* out, int din, int dout, int batch) {
        k_transpose_bf16<<<div_up((size_t)batch * din * dout, 256), 256, 0, stream>>>(in, out, din, dout, batch);
    };

    // --- input linear + relu ---------------------------------------------
    T(Pf(63), lt_a, 128, 256, 1);
    T(Pf(65), lt_p, 128, 256, 1);
    k_to_bf16<<<div_up((size_t)NA * 128, 256), 256, 0, stream>>>(xa, xb_a, NA * 128);
    k_to_bf16<<<div_up((size_t)NP * 128, 256), 256, 0, stream>>>(xp, xb_p, NP * 128);
    gemm(xb_a, lt_a, Pf(64), h_a, nullptr, NA, 128, 256, 1);
    gemm(xb_p, lt_p, Pf(66), h_p, nullptr, NP, 128, 256, 1);

    // --- HGT layers -------------------------------------------------------
    for (int l = 0; l < 2; ++l) {
        int B = 5 + l * 29;
        // per-layer weight prep (transposed bf16)
        T(Pf(B + 5),  kt_a, 256, 256, 1); T(Pf(B + 7),  kt_p, 256, 256, 1);
        T(Pf(B + 19), qt_a, 256, 256, 1); T(Pf(B + 21), qt_p, 256, 256, 1);
        T(Pf(B + 25), vt_a, 256, 256, 1); T(Pf(B + 27), vt_p, 256, 256, 1);
        T(Pf(B + 12), ot_a, 256, 256, 1); T(Pf(B + 14), ot_p, 256, 256, 1);
        // relation order: {writes, rev_writes, cites}
        int aIdx[3] = {B + 2, B + 1, B + 0};
        int mIdx[3] = {B + 11, B + 10, B + 9};
        int pIdx[3] = {B + 18, B + 17, B + 16};
        for (int r = 0; r < 3; ++r) { T(Pf(aIdx[r]), art[r], 32, 32, 8);
                                      T(Pf(mIdx[r]), mrt[r], 32, 32, 8); }

        k_to_bf16<<<div_up((size_t)NA * 256, 256), 256, 0, stream>>>(h_a, hb_a, NA * 256);
        k_to_bf16<<<div_up((size_t)NP * 256, 256), 256, 0, stream>>>(h_p, hb_p, NP * 256);

        // K / V projections straight to bf16; Q kept f32 for the alpha dot
        gemm(hb_a, kt_a, Pf(B + 6),  nullptr, kb_a, NA, 256, 256, 0);
        gemm(hb_p, kt_p, Pf(B + 8),  nullptr, kb_p, NP, 256, 256, 0);
        gemm(hb_a, vt_a, Pf(B + 26), nullptr, vb_a, NA, 256, 256, 0);
        gemm(hb_p, vt_p, Pf(B + 28), nullptr, vb_p, NP, 256, 256, 0);
        gemm(hb_a, qt_a, Pf(B + 20), q_a, nullptr, NA, 256, 256, 0);
        gemm(hb_p, qt_p, Pf(B + 22), q_p, nullptr, NP, 256, 256, 0);

        // init segment buffers
        k_fill<<<div_up((size_t)NA * 8, 256), 256, 0, stream>>>(m_a, -INFINITY, NA * 8);
        k_fill<<<div_up((size_t)NP * 8, 256), 256, 0, stream>>>(m_p, -INFINITY, NP * 8);
        k_fill<<<div_up((size_t)NA * 8, 256), 256, 0, stream>>>(s_a, 0.0f, NA * 8);
        k_fill<<<div_up((size_t)NP * 8, 256), 256, 0, stream>>>(s_p, 0.0f, NP * 8);
        k_fill<<<div_up((size_t)NA * 256, 256), 256, 0, stream>>>(agg_a, 0.0f, NA * 256);
        k_fill<<<div_up((size_t)NP * 256, 256), 256, 0, stream>>>(agg_p, 0.0f, NP * 256);

        const int* eptr[3] = {ew, erw, ec};
        int Ecnt[3] = {E_W, E_RW, E_C};
        int off[3]  = {0, E_W, E_W + E_RW};
        const unsigned short* ksrc[3] = {kb_a, kb_p, kb_p};
        const unsigned short* vsrc[3] = {vb_a, vb_p, vb_p};
        const float* qdst[3] = {q_p, q_a, q_p};
        float* mdst[3] = {m_p, m_a, m_p};
        float* sdst[3] = {s_p, s_a, s_p};
        float* adst[3] = {agg_p, agg_a, agg_p};

        for (int r = 0; r < 3; ++r) {
            const int* src = eptr[r];
            const int* dst = eptr[r] + Ecnt[r];
            int E = Ecnt[r];
            unsigned eblk = div_up((size_t)(E / 16) * NHEADS, 8);
            k_edge<<<eblk, 256, 0, stream>>>(ksrc[r], src, art[r], keb, E);
            k_edge<<<eblk, 256, 0, stream>>>(vsrc[r], src, mrt[r],
                                             veb + (size_t)off[r] * HIDDEN, E);
            k_alpha<<<div_up((size_t)E * 8, 256), 256, 0, stream>>>(
                keb, qdst[r], dst, Pf(pIdx[r]), alpha + (size_t)off[r] * 8, mdst[r], E);
        }
        for (int r = 0; r < 3; ++r) {
            const int* dst = eptr[r] + Ecnt[r];
            int E = Ecnt[r];
            k_expsum<<<div_up((size_t)E * 8, 256), 256, 0, stream>>>(
                mdst[r], dst, alpha + (size_t)off[r] * 8, sdst[r], E);
        }
        for (int r = 0; r < 3; ++r) {
            const int* dst = eptr[r] + Ecnt[r];
            int E = Ecnt[r];
            k_scatter<<<div_up((size_t)E * 8, 256), 256, 0, stream>>>(
                veb + (size_t)off[r] * HIDDEN, dst, alpha + (size_t)off[r] * 8,
                sdst[r], adst[r], E);
        }

        // GELU -> out projection -> skip mix -> BatchNorm
        k_gelu_bf16<<<div_up((size_t)NA * 256, 256), 256, 0, stream>>>(agg_a, gb_a, NA * 256);
        k_gelu_bf16<<<div_up((size_t)NP * 256, 256), 256, 0, stream>>>(agg_p, gb_p, NP * 256);
        gemm(gb_a, ot_a, Pf(B + 13), o_a, nullptr, NA, 256, 256, 0);
        gemm(gb_p, ot_p, Pf(B + 15), o_p, nullptr, NP, 256, 256, 0);
        k_mix<<<div_up((size_t)NA * 256, 256), 256, 0, stream>>>(o_a, Pf(B + 23), h_a, NA * 256);
        k_mix<<<div_up((size_t)NP * 256, 256), 256, 0, stream>>>(o_p, Pf(B + 24), h_p, NP * 256);

        k_bn_stats<<<256, 256, 0, stream>>>(h_a, NA, bns);
        k_bn_apply<<<div_up((size_t)NA * 256, 256), 256, 0, stream>>>(h_a, bns, Pf(B + 4), Pf(B + 3), NA);
        k_bn_stats<<<256, 256, 0, stream>>>(h_p, NP, bns);
        k_bn_apply<<<div_up((size_t)NP * 256, 256), 256, 0, stream>>>(h_p, bns, Pf(B + 4), Pf(B + 3), NP);
    }

    // --- output: author block then paper block ---------------------------
    float* out = (float*)d_out;
    k_copy<<<div_up((size_t)NA * 256, 256), 256, 0, stream>>>(h_a, out, NA * 256);
    k_copy<<<div_up((size_t)NP * 256, 256), 256, 0, stream>>>(h_p, out + (size_t)NA * 256, NP * 256);
}
